// Encoder_80874234183807
// MI455X (gfx1250) — matrix operand order repaired and checked by reading, not compiled, not run on hardware
//
#include <hip/hip_runtime.h>

// ---------------------------------------------------------------------------
// CDNA5 (gfx1250, wave32) encoder: BiLSTM stack + AST LSTM + dense merges.
// Heavy path: v_wmma_f32_16x16x32_bf16 on [16,512]x[512,1024] per-step GEMMs.
// ---------------------------------------------------------------------------

typedef __attribute__((ext_vector_type(16))) __bf16 v16bf;
typedef __attribute__((ext_vector_type(8)))  float  v8f;

union BF16Frag {
    struct Pair { uint4 lo, hi; } u;   // 2 x 16B = 32B = 16 bf16
    v16bf v;
};

#define WRSZ (512 * 1024)   // bf16 elements per repacked weight set
#define XH_STRIDE 520       // 512 + 8 pad halves -> conflict-free b128 LDS loads

__device__ __forceinline__ unsigned short f32_to_bf16(float f) {
    unsigned int u = __float_as_uint(f);
    u += 0x7FFFu + ((u >> 16) & 1u);   // round-to-nearest-even
    return (unsigned short)(u >> 16);
}
__device__ __forceinline__ unsigned int pack_bf16x2(float a, float b) {
    return (unsigned int)f32_to_bf16(a) | ((unsigned int)f32_to_bf16(b) << 16);
}
__device__ __forceinline__ float sigm(float x) {
    return 1.0f / (1.0f + __expf(-x));
}

// ---------------------------------------------------------------------------
// 1) Repack fp32 W[E,4H] / R[H,4H] into bf16 WMMA B-fragment layout.
//    Fragment element e (0..15) of lane L in k-tile kk, n-tile nt holds
//    B[K][N] with N = nt*16 + (L%16), K = kk*32 + e + (L/16)*16.
// ---------------------------------------------------------------------------
struct RepackP {
    const float* W[7];
    const float* R[7];
    unsigned short* out;
};

__global__ __launch_bounds__(256) void repack_weights(RepackP P) {
    int set = blockIdx.x >> 11;                        // / 2048
    int o   = ((blockIdx.x & 2047) << 8) | threadIdx.x; // 0 .. 512*1024-1
    int e    =  o        & 15;
    int lane = (o >> 4)  & 31;
    int nt   = (o >> 9)  & 63;
    int kk   =  o >> 15;
    int g    = lane >> 4;
    int K = kk * 32 + e + g * 16;
    int N = nt * 16 + (lane & 15);
    float v = (K < 256) ? P.W[set][K * 1024 + N]
                        : P.R[set][(K - 256) * 1024 + N];
    P.out[(size_t)set * WRSZ + o] = f32_to_bf16(v);
}

// ---------------------------------------------------------------------------
// 2) Mega LSTM kernel: 26 workgroups cover every encoder/direction.
//    Each WG owns 16 sequences; 16 waves each own one 16-col slice of all
//    four gates (tiles G*16 + wave), so gate fusion is wave-local.
// ---------------------------------------------------------------------------
struct LstmJobs {
    const int* cm_tok; const int* sc_tok; const int* iss_tok;
    const int* old_tok; const int* cur_tok;
    const float* emb_c; const float* emb_s; const float* emb_i; const float* emb_a;
    const unsigned short* wr;                 // 7 repacked weight sets
    const float* cb; const float* sb; const float* ib; const float* ab;
    float* h_cm; float* c_cm; float* h_sc; float* c_sc;
    float* h_if; float* c_if; float* h_ib; float* c_ib;
    float* h_ast; float* c_ast;
};

__global__ __launch_bounds__(512) void lstm_all(LstmJobs P) {
    __shared__ __align__(16) unsigned short xh[16][XH_STRIDE]; // [seq][x(256)|h(256)] bf16

    const int bid = blockIdx.x;
    const int* tok; const float* emb; const unsigned short* WR; const float* bias;
    float* hout; float* cout;
    int T, nseq, rev, ostr, ocol, blk;

    if (bid < 2)       { blk = bid;      tok = P.cm_tok;  T = 64;  nseq = 32;  rev = 0; emb = P.emb_c; WR = P.wr + 0 * WRSZ; bias = P.cb;        hout = P.h_cm;  cout = P.c_cm;  ostr = 512; ocol = 0;   }
    else if (bid < 4)  { blk = bid - 2;  tok = P.cm_tok;  T = 64;  nseq = 32;  rev = 1; emb = P.emb_c; WR = P.wr + 1 * WRSZ; bias = P.cb + 1024; hout = P.h_cm;  cout = P.c_cm;  ostr = 512; ocol = 256; }
    else if (bid < 6)  { blk = bid - 4;  tok = P.sc_tok;  T = 128; nseq = 32;  rev = 0; emb = P.emb_s; WR = P.wr + 2 * WRSZ; bias = P.sb;        hout = P.h_sc;  cout = P.c_sc;  ostr = 512; ocol = 0;   }
    else if (bid < 8)  { blk = bid - 6;  tok = P.sc_tok;  T = 128; nseq = 32;  rev = 1; emb = P.emb_s; WR = P.wr + 3 * WRSZ; bias = P.sb + 1024; hout = P.h_sc;  cout = P.c_sc;  ostr = 512; ocol = 256; }
    else if (bid == 8) { blk = 0;        tok = P.iss_tok; T = 32;  nseq = 4;   rev = 0; emb = P.emb_i; WR = P.wr + 4 * WRSZ; bias = P.ib;        hout = P.h_if;  cout = P.c_if;  ostr = 256; ocol = 0;   }
    else if (bid == 9) { blk = 0;        tok = P.iss_tok; T = 32;  nseq = 4;   rev = 1; emb = P.emb_i; WR = P.wr + 5 * WRSZ; bias = P.ib + 1024; hout = P.h_ib;  cout = P.c_ib;  ostr = 256; ocol = 0;   }
    else if (bid < 18) { blk = bid - 10; tok = P.old_tok; T = 256; nseq = 128; rev = 0; emb = P.emb_a; WR = P.wr + 6 * WRSZ; bias = P.ab;        hout = P.h_ast; cout = P.c_ast; ostr = 512; ocol = 0;   }
    else               { blk = bid - 18; tok = P.cur_tok; T = 256; nseq = 128; rev = 0; emb = P.emb_a; WR = P.wr + 6 * WRSZ; bias = P.ab;        hout = P.h_ast; cout = P.c_ast; ostr = 512; ocol = 256; }

    const int tid  = threadIdx.x;
    const int lane = tid & 31;
    const int wave = tid >> 5;       // 0..15
    const int lm   = lane & 15;
    const int g    = lane >> 4;      // lane half (C/D row-group select)
    const int j    = wave * 16 + lm; // hidden column owned by this lane

    // Per-lane gate biases (gate order i,f,g,o along 4H).
    const float bI = bias[j], bF = bias[256 + j], bG = bias[512 + j], bO = bias[768 + j];

    // Zero the h half of the LDS staging buffer (h_0 = 0).
    {
        int s   = tid >> 5;
        int col = 256 + ((tid & 31) << 3);
        *(uint4*)(&xh[s][col]) = make_uint4(0u, 0u, 0u, 0u);
    }
    float creg[8];
#pragma unroll
    for (int r = 0; r < 8; ++r) creg[r] = 0.0f;

    const uint4* __restrict__ wq = (const uint4*)WR;

    for (int t = 0; t < T; ++t) {
        const int tt = rev ? (T - 1 - t) : t;

        // --- gather embeddings -> bf16 into xh[s][0:256] -------------------
        {
            int s  = tid >> 5;           // sequence within group
            int d0 = (tid & 31) << 3;    // 8 floats per thread
            int sg = blk * 16 + s;
            int tk = (sg < nseq) ? tok[sg * T + tt] : 0;
            const float* er = emb + (size_t)tk * 256 + d0;
            uint4 pk;
            pk.x = pack_bf16x2(er[0], er[1]);
            pk.y = pack_bf16x2(er[2], er[3]);
            pk.z = pack_bf16x2(er[4], er[5]);
            pk.w = pack_bf16x2(er[6], er[7]);
            *(uint4*)(&xh[s][d0]) = pk;
        }
        __syncthreads();   // gather + previous step's h writes visible

        // --- z = [x|h] @ [W;R] : 16 k-steps, 4 gate tiles per wave ---------
        v8f acc[4];
#pragma unroll
        for (int G = 0; G < 4; ++G) { v8f zz = {}; acc[G] = zz; }

        for (int kk = 0; kk < 16; ++kk) {
            // A fragment (16x32 bf16): lane lm = row M; two 16B LDS chunks.
            BF16Frag a;
            const unsigned short* rp = &xh[lm][kk * 32 + g * 16];
            a.u.lo = *(const uint4*)rp;
            a.u.hi = *(const uint4*)(rp + 8);
#pragma unroll
            for (int G = 0; G < 4; ++G) {
                const int nt = G * 16 + wave;
                const int fi = ((kk * 64 + nt) * 32 + lane) * 2;  // uint4 units
                BF16Frag b;
                b.u.lo = wq[fi];
                b.u.hi = wq[fi + 1];
                acc[G] = __builtin_amdgcn_wmma_f32_16x16x32_bf16(
                    false, a.v, false, b.v, (short)0, acc[G], false, false);
            }
        }
        __syncthreads();   // all waves done reading xh before h is rewritten

        // --- gates + state update (all wave-local) -------------------------
        const bool last = (t == T - 1);
#pragma unroll
        for (int r = 0; r < 8; ++r) {
            const int M = r + 8 * g;   // C/D layout: row = r + 8*(lane/16)
            float iv = sigm(acc[0][r] + bI);
            float fv = sigm(acc[1][r] + bF);
            float gv = tanhf(acc[2][r] + bG);
            float ov = sigm(acc[3][r] + bO);
            float cn = fv * creg[r] + iv * gv;
            float hn = ov * tanhf(cn);
            creg[r] = cn;
            xh[M][256 + j] = f32_to_bf16(hn);
            if (last && (blk * 16 + M) < nseq) {
                size_t o = (size_t)(blk * 16 + M) * ostr + ocol + j;
                hout[o] = hn;
                cout[o] = cn;
            }
        }
    }
}

// ---------------------------------------------------------------------------
// 3) AST merge: h_ast = concat(ho,hn)@W_mah+b_mah; h_asts = h_ast@W_mall+b.
// ---------------------------------------------------------------------------
__global__ __launch_bounds__(256) void ast_merge(
    const float* __restrict__ hcat, const float* __restrict__ ccat,
    const float* __restrict__ W_mah, const float* __restrict__ b_mah,
    const float* __restrict__ W_mac, const float* __restrict__ b_mac,
    const float* __restrict__ W_mall, const float* __restrict__ b_mall,
    float* __restrict__ h_asts, float* __restrict__ c_asts) {
    const int row = blockIdx.x, tid = threadIdx.x;
    const float* hr = hcat + row * 512;
    const float* cr = ccat + row * 512;
    float hv = b_mah[tid], cv = b_mac[tid];
    for (int k = 0; k < 512; ++k) {
        hv += hr[k] * W_mah[k * 256 + tid];
        cv += cr[k] * W_mac[k * 256 + tid];
    }
    __shared__ float red[2][256];
    red[0][tid] = hv * W_mall[tid];
    red[1][tid] = cv * W_mall[tid];
    __syncthreads();
    for (int s = 128; s > 0; s >>= 1) {
        if (tid < s) { red[0][tid] += red[0][tid + s]; red[1][tid] += red[1][tid + s]; }
        __syncthreads();
    }
    if (tid == 0) {
        h_asts[row] = red[0][0] + b_mall[0];
        c_asts[row] = red[1][0] + b_mall[0];
    }
}

// ---------------------------------------------------------------------------
// 4) Per-commit merge: dot(concat(h_cm,h_sc,h_asts)[1028], W_mcom) + b.
// ---------------------------------------------------------------------------
__global__ __launch_bounds__(256) void commit_merge(
    const float* __restrict__ h_cm, const float* __restrict__ c_cm,
    const float* __restrict__ h_sc, const float* __restrict__ c_sc,
    const float* __restrict__ h_asts, const float* __restrict__ c_asts,
    const float* __restrict__ W_mcom, const float* __restrict__ b_mcom,
    float* __restrict__ h_com, float* __restrict__ c_com) {
    const int row = blockIdx.x, tid = threadIdx.x;
    float hv = 0.f, cv = 0.f;
    for (int e = tid; e < 1028; e += 256) {
        float he, ce;
        if (e < 512)       { he = h_cm[row * 512 + e];          ce = c_cm[row * 512 + e]; }
        else if (e < 1024) { he = h_sc[row * 512 + e - 512];    ce = c_sc[row * 512 + e - 512]; }
        else               { he = h_asts[row * 4 + e - 1024];   ce = c_asts[row * 4 + e - 1024]; }
        hv += he * W_mcom[e];
        cv += ce * W_mcom[e];
    }
    __shared__ float red[2][256];
    red[0][tid] = hv; red[1][tid] = cv;
    __syncthreads();
    for (int s = 128; s > 0; s >>= 1) {
        if (tid < s) { red[0][tid] += red[0][tid + s]; red[1][tid] += red[1][tid + s]; }
        __syncthreads();
    }
    if (tid == 0) {
        h_com[row] = red[0][0] + b_mcom[0];
        c_com[row] = red[1][0] + b_mcom[0];
    }
}

// ---------------------------------------------------------------------------
// 5) Final merge: out = stack([concat(h_commits, hf+hb) @ W_mh + b_mh, ...c])
// ---------------------------------------------------------------------------
__global__ __launch_bounds__(256) void final_merge(
    const float* __restrict__ h_com, const float* __restrict__ c_com,
    const float* __restrict__ h_if, const float* __restrict__ c_if,
    const float* __restrict__ h_ib, const float* __restrict__ c_ib,
    const float* __restrict__ W_mh, const float* __restrict__ b_mh,
    const float* __restrict__ W_mc, const float* __restrict__ b_mc,
    float* __restrict__ out) {
    const int b = blockIdx.x, j = threadIdx.x;
    float h = b_mh[j], c = b_mc[j];
    for (int k = 0; k < 8; ++k) {
        h += h_com[b * 8 + k] * W_mh[k * 256 + j];
        c += c_com[b * 8 + k] * W_mc[k * 256 + j];
    }
    for (int k = 0; k < 256; ++k) {
        h += (h_if[b * 256 + k] + h_ib[b * 256 + k]) * W_mh[(8 + k) * 256 + j];
        c += (c_if[b * 256 + k] + c_ib[b * 256 + k]) * W_mc[(8 + k) * 256 + j];
    }
    out[b * 256 + j]        = h;
    out[1024 + b * 256 + j] = c;
}

// ---------------------------------------------------------------------------
extern "C" void kernel_launch(void* const* d_in, const int* in_sizes, int n_in,
                              void* d_out, int out_size, void* d_ws, size_t ws_size,
                              hipStream_t stream) {
    (void)in_sizes; (void)n_in; (void)out_size; (void)ws_size;

    const int*   cm_tok  = (const int*)d_in[0];
    const int*   sc_tok  = (const int*)d_in[1];
    const int*   old_tok = (const int*)d_in[2];
    const int*   cur_tok = (const int*)d_in[3];
    const int*   iss_tok = (const int*)d_in[4];
    const float* emb_c   = (const float*)d_in[5];
    const float* emb_s   = (const float*)d_in[6];
    const float* emb_i   = (const float*)d_in[7];
    const float* emb_a   = (const float*)d_in[8];
    const float* cW = (const float*)d_in[9];
    const float* cR = (const float*)d_in[10];
    const float* cb = (const float*)d_in[11];
    const float* sW = (const float*)d_in[12];
    const float* sR = (const float*)d_in[13];
    const float* sb = (const float*)d_in[14];
    const float* iW = (const float*)d_in[15];
    const float* iR = (const float*)d_in[16];
    const float* ib = (const float*)d_in[17];
    const float* aW = (const float*)d_in[18];
    const float* aR = (const float*)d_in[19];
    const float* ab = (const float*)d_in[20];
    const float* W_mah  = (const float*)d_in[21];
    const float* b_mah  = (const float*)d_in[22];
    const float* W_mac  = (const float*)d_in[23];
    const float* b_mac  = (const float*)d_in[24];
    const float* W_mall = (const float*)d_in[25];
    const float* b_mall = (const float*)d_in[26];
    const float* W_mcom = (const float*)d_in[27];
    const float* b_mcom = (const float*)d_in[28];
    const float* W_mh   = (const float*)d_in[29];
    const float* b_mh   = (const float*)d_in[30];
    const float* W_mc   = (const float*)d_in[31];
    const float* b_mc   = (const float*)d_in[32];

    // ---- workspace layout (~8.2 MB) ----
    char* ws = (char*)d_ws;
    unsigned short* wr = (unsigned short*)ws;              // 7 * WRSZ bf16 = 7 MiB
    float* h_cm  = (float*)(ws + (size_t)7 * WRSZ * 2);    // [32][512]
    float* c_cm  = h_cm  + 32 * 512;
    float* h_sc  = c_cm  + 32 * 512;
    float* c_sc  = h_sc  + 32 * 512;
    float* h_if  = c_sc  + 32 * 512;                       // [4][256]
    float* c_if  = h_if  + 4 * 256;
    float* h_ib  = c_if  + 4 * 256;
    float* c_ib  = h_ib  + 4 * 256;
    float* h_ast = c_ib  + 4 * 256;                        // [128][512]
    float* c_ast = h_ast + 128 * 512;
    float* h_asts = c_ast + 128 * 512;                     // [128]
    float* c_asts = h_asts + 128;
    float* h_com  = c_asts + 128;                          // [32]
    float* c_com  = h_com + 32;

    // ---- 1) repack weights to bf16 WMMA fragment layout ----
    RepackP rp;
    rp.W[0] = cW;          rp.R[0] = cR;
    rp.W[1] = cW + 262144; rp.R[1] = cR + 262144;
    rp.W[2] = sW;          rp.R[2] = sR;
    rp.W[3] = sW + 262144; rp.R[3] = sR + 262144;
    rp.W[4] = iW;          rp.R[4] = iR;
    rp.W[5] = iW + 262144; rp.R[5] = iR + 262144;
    rp.W[6] = aW;          rp.R[6] = aR;
    rp.out = wr;
    repack_weights<<<7 * 2048, 256, 0, stream>>>(rp);

    // ---- 2) all LSTMs in one launch (26 independent workgroups) ----
    LstmJobs J;
    J.cm_tok = cm_tok; J.sc_tok = sc_tok; J.iss_tok = iss_tok;
    J.old_tok = old_tok; J.cur_tok = cur_tok;
    J.emb_c = emb_c; J.emb_s = emb_s; J.emb_i = emb_i; J.emb_a = emb_a;
    J.wr = wr;
    J.cb = cb; J.sb = sb; J.ib = ib; J.ab = ab;
    J.h_cm = h_cm; J.c_cm = c_cm; J.h_sc = h_sc; J.c_sc = c_sc;
    J.h_if = h_if; J.c_if = c_if; J.h_ib = h_ib; J.c_ib = c_ib;
    J.h_ast = h_ast; J.c_ast = c_ast;
    lstm_all<<<26, 512, 0, stream>>>(J);

    // ---- 3..5) dense merges ----
    ast_merge<<<128, 256, 0, stream>>>(h_ast, c_ast, W_mah, b_mah, W_mac, b_mac,
                                       W_mall, b_mall, h_asts, c_asts);
    commit_merge<<<32, 256, 0, stream>>>(h_cm, c_cm, h_sc, c_sc, h_asts, c_asts,
                                         W_mcom, b_mcom, h_com, c_com);
    final_merge<<<4, 256, 0, stream>>>(h_com, c_com, h_if, c_if, h_ib, c_ib,
                                       W_mh, b_mh, W_mc, b_mc, (float*)d_out);
}
